// HeteroGraphHAN_70300024700960
// MI455X (gfx1250) — matrix operand order repaired and checked by reading, not compiled, not run on hardware
//
#include <hip/hip_runtime.h>
#include <hip/hip_bf16.h>
#include <cstdint>
#include <cstddef>

// Problem constants (match reference)
#define NN 100000   // operator nodes
#define CC 256      // channels
#define PP 10       // meta-paths
#define EE 500000   // edges per path
#define BB 64       // batch
#define HH 4        // heads (DH = 64)

typedef __attribute__((ext_vector_type(16))) __bf16 v16bf;
typedef __attribute__((ext_vector_type(8)))  float  v8f;

// ---------------- order-preserving float<->uint keys for atomic max ----------
__device__ __forceinline__ unsigned fkey(float x) {
  unsigned u = __float_as_uint(x);
  return (u & 0x80000000u) ? ~u : (u | 0x80000000u);
}
__device__ __forceinline__ float fdec(unsigned k) {
  unsigned u = (k & 0x80000000u) ? (k & 0x7FFFFFFFu) : ~k;
  return __uint_as_float(u);
}

// ---------------- WMMA bf16 16x16x32, 16x64 register-blocked wave tile -------
// A: [M, 256] row-major bf16. B: [256, 256] row-major bf16 (K rows, N cols).
// A-frag: lane%16 = M row; lane/16 selects K octet; VGPR v packs K pairs.
// B-frag: lane%16 = N col; lane/16 selects K octet; VGPR v packs K pairs (as A),
//         n-subtile j reads cols n0 + 16*j; each element comes from a CC-strided K row.
// One A fragment is reused by 4 WMMAs (4x less A traffic vs 16x16 tiling).
__device__ __forceinline__ void wmma_tile_16x64(const __bf16* __restrict__ A,
                                                const __bf16* __restrict__ B,
                                                int m0, int n0, int lane,
                                                v8f acc[4]) {
  const int mrow = m0 + (lane & 15);
  const int kgrp = lane >> 4;
  for (int k0 = 0; k0 < CC; k0 += 32) {
    v16bf af;
    const __bf16* ap = A + (size_t)mrow * CC + k0 + (kgrp << 3);
#pragma unroll
    for (int v = 0; v < 8; ++v) {
      const int kk = ((v & 4) << 2) + ((v & 3) << 1);  // 0,2,4,6,16,18,20,22
      af[2 * v]     = ap[kk];
      af[2 * v + 1] = ap[kk + 1];
    }
    const __bf16* bp = B + (size_t)(k0 + (kgrp << 3)) * CC + n0 + (lane & 15);
#pragma unroll
    for (int j = 0; j < 4; ++j) {
      v16bf bfr;
#pragma unroll
      for (int v = 0; v < 8; ++v) {
        const int kk = ((v & 4) << 2) + ((v & 3) << 1);
        bfr[2 * v]     = bp[(size_t)kk * CC + 16 * j];
        bfr[2 * v + 1] = bp[(size_t)(kk + 1) * CC + 16 * j];
      }
      acc[j] = __builtin_amdgcn_wmma_f32_16x16x32_bf16(
          false, af, false, bfr, (short)0, acc[j], false, false);
    }
  }
}

// C/D layout: value (m0 + v + 8*(lane>>4), n0 + 16*j + lane&15) in acc[j][v]
__global__ __launch_bounds__(256) void wmma_gemm_bias(
    const __bf16* __restrict__ A, const __bf16* __restrict__ B,
    const float* __restrict__ bias, float* __restrict__ C, int wtiles) {
  const int wid  = (blockIdx.x * blockDim.x + threadIdx.x) >> 5;
  const int lane = threadIdx.x & 31;
  if (wid >= wtiles) return;
  const int m0 = (wid >> 2) << 4;   // CC/64 = 4 n-groups per m-tile
  const int n0 = (wid & 3) << 6;
  v8f acc[4] = {{}, {}, {}, {}};
  wmma_tile_16x64(A, B, m0, n0, lane, acc);
  const int cmb = m0 + ((lane >> 4) << 3);
#pragma unroll
  for (int j = 0; j < 4; ++j) {
    const int cn = n0 + 16 * j + (lane & 15);
    const float bv = bias[cn];
#pragma unroll
    for (int v = 0; v < 8; ++v) C[(size_t)(cmb + v) * CC + cn] = acc[j][v] + bv;
  }
}

// z @ kW with fused epilogue: wbar[p] += sum tanh(t + kb[c]) * q[c]
// 100000 % 16 == 0 => every 16-row tile belongs to exactly one meta-path p.
__global__ __launch_bounds__(256) void wmma_gemm_tanhq(
    const __bf16* __restrict__ A, const __bf16* __restrict__ B,
    const float* __restrict__ kb, const float* __restrict__ q,
    float* __restrict__ wbar, int wtiles) {
  const int wid  = (blockIdx.x * blockDim.x + threadIdx.x) >> 5;
  const int lane = threadIdx.x & 31;
  if (wid >= wtiles) return;
  const int m0 = (wid >> 2) << 4;
  const int n0 = (wid & 3) << 6;
  v8f acc[4] = {{}, {}, {}, {}};
  wmma_tile_16x64(A, B, m0, n0, lane, acc);
  float part = 0.f;
#pragma unroll
  for (int j = 0; j < 4; ++j) {
    const int cn = n0 + 16 * j + (lane & 15);
    const float kbv = kb[cn], qv = q[cn];
#pragma unroll
    for (int v = 0; v < 8; ++v) part += tanhf(acc[j][v] + kbv) * qv;
  }
#pragma unroll
  for (int off = 16; off > 0; off >>= 1) part += __shfl_down(part, off, 32);
  if (lane == 0) atomicAdd(&wbar[m0 / NN], part);
}

// ---------------- small elementwise / setup kernels --------------------------
__global__ __launch_bounds__(256) void proj_op_kernel(
    const float* __restrict__ x, const float* __restrict__ W,
    const float* __restrict__ b, float* __restrict__ h) {
  const size_t i = (size_t)blockIdx.x * 256 + threadIdx.x;  // over NN*CC
  const int n = (int)(i >> 8), c = (int)(i & 255);
  const float* xr = x + (size_t)n * 4;
  float s = b[c];
#pragma unroll
  for (int k = 0; k < 4; ++k) s += xr[k] * W[k * CC + c];
  h[i] = s;
}

__global__ __launch_bounds__(256) void f32_to_bf16_kernel(
    const float* __restrict__ in, __bf16* __restrict__ out, size_t n) {
  const size_t i = (size_t)blockIdx.x * 256 + threadIdx.x;
  if (i < n) out[i] = (__bf16)in[i];
}

// a_src[p][n][h], a_dst[p][n][h] : 64-length head dots against att params.
// The 1KB h-row is staged into LDS with the CDNA5 async direct-to-LDS path
// (ASYNCcnt-tracked, no VGPR round trip).
__global__ __launch_bounds__(128) void att_kernel(
    const float* __restrict__ h, const float* __restrict__ att_src,
    const float* __restrict__ att_dst, float* __restrict__ a_src,
    float* __restrict__ a_dst) {
  __shared__ float hn[CC];
  const int n = blockIdx.x;
  const int t = threadIdx.x;
  if (t < 64) {  // 64 lanes x 16B = 1KB row
    const float* gsrc = h + (size_t)n * CC + t * 4;
    unsigned lds_off = (unsigned)(reinterpret_cast<uintptr_t>(&hn[t * 4]));
    asm volatile("global_load_async_to_lds_b128 %0, %1, off"
                 :: "v"(lds_off), "v"(gsrc) : "memory");
  }
  asm volatile("s_wait_asynccnt 0x0" ::: "memory");
  __syncthreads();
  if (t < 2 * PP * HH) {
    const int which = t / (PP * HH);
    const int ph = t % (PP * HH);
    const int p = ph / HH, hd = ph % HH;
    const float* att = (which ? att_dst : att_src) + (size_t)(p * HH + hd) * 64;
    const float* hv = hn + hd * 64;
    float s = 0.f;
#pragma unroll
    for (int d = 0; d < 64; ++d) s += hv[d] * att[d];
    float* out = which ? a_dst : a_src;
    out[((size_t)p * NN + n) * HH + hd] = s;
  }
}

__global__ __launch_bounds__(256) void path_init_kernel(
    float* __restrict__ outacc, unsigned* __restrict__ mkey,
    float* __restrict__ denom) {
  const size_t i = (size_t)blockIdx.x * 256 + threadIdx.x;  // NN*CC threads
  outacc[i] = 0.f;
  if (i < (size_t)NN * HH) {
    mkey[i]  = 0x007FFFFFu;  // fkey(-inf)
    denom[i] = 0.f;
  }
}

__global__ __launch_bounds__(256) void edge_pass1(
    const int* __restrict__ src, const int* __restrict__ dst,
    const float* __restrict__ asp, const float* __restrict__ adp,
    float* __restrict__ e_store, unsigned* __restrict__ mkey) {
  const int eid = blockIdx.x * 256 + threadIdx.x;
  if (eid >= EE) return;
  const int s = src[eid], d = dst[eid];
#pragma unroll
  for (int hd = 0; hd < HH; ++hd) {
    float e = asp[(size_t)s * HH + hd] + adp[(size_t)d * HH + hd];
    e = e > 0.f ? e : 0.2f * e;  // leaky_relu
    e_store[(size_t)eid * HH + hd] = e;
    atomicMax(&mkey[d * HH + hd], fkey(e));
  }
}

__global__ __launch_bounds__(256) void edge_pass2(
    const int* __restrict__ dst, float* __restrict__ e_store,
    const unsigned* __restrict__ mkey, float* __restrict__ denom) {
  const int eid = blockIdx.x * 256 + threadIdx.x;
  if (eid >= EE) return;
  const int d = dst[eid];
#pragma unroll
  for (int hd = 0; hd < HH; ++hd) {
    const float m = fdec(mkey[d * HH + hd]);
    const float ex = __expf(e_store[(size_t)eid * HH + hd] - m);
    e_store[(size_t)eid * HH + hd] = ex;  // overwrite with exp
    atomicAdd(&denom[d * HH + hd], ex);
  }
}

// one block per edge, one thread per channel; h and outacc are L2-resident
__global__ __launch_bounds__(256) void edge_pass3(
    const int* __restrict__ src, const int* __restrict__ dst,
    const float* __restrict__ e_store, const float* __restrict__ denom,
    const float* __restrict__ hW, float* __restrict__ outacc) {
  const int eid = blockIdx.x;
  const int c = threadIdx.x;
  const int hd = c >> 6;
  const int s = src[eid], d = dst[eid];
  const float alpha = e_store[(size_t)eid * HH + hd] / denom[d * HH + hd];
  atomicAdd(&outacc[(size_t)d * CC + c], hW[(size_t)s * CC + c] * alpha);
}

__global__ __launch_bounds__(256) void relu_cvt_kernel(
    const float* __restrict__ in, __bf16* __restrict__ z) {
  const size_t i = (size_t)blockIdx.x * 256 + threadIdx.x;  // NN*CC threads
  const float v = in[i];
  z[i] = (__bf16)(v > 0.f ? v : 0.f);
}

__global__ void zero_f_kernel(float* __restrict__ p, int n) {
  const int i = blockIdx.x * 256 + threadIdx.x;
  if (i < n) p[i] = 0.f;
}

__global__ void beta_kernel(const float* __restrict__ wbar,
                            float* __restrict__ beta) {
  if (threadIdx.x == 0) {
    float w[PP];
    float mx = -3.4e38f;
#pragma unroll
    for (int p = 0; p < PP; ++p) {
      w[p] = wbar[p] * (1.f / (float)NN);
      mx = fmaxf(mx, w[p]);
    }
    float s = 0.f;
#pragma unroll
    for (int p = 0; p < PP; ++p) {
      w[p] = __expf(w[p] - mx);
      s += w[p];
    }
#pragma unroll
    for (int p = 0; p < PP; ++p) beta[p] = w[p] / s;
  }
}

// x[n,c] = layer_norm(elu(sum_p beta[p] * z[p,n,c])) ; block = one node
__global__ __launch_bounds__(256) void combine_elu_norm(
    const __bf16* __restrict__ z, const float* __restrict__ beta,
    const float* __restrict__ g, const float* __restrict__ b,
    float* __restrict__ xout) {
  __shared__ float red[CC];
  __shared__ float s_mu, s_var;
  const int n = blockIdx.x, c = threadIdx.x;
  const size_t idx = (size_t)n * CC + c;
  float v = 0.f;
#pragma unroll
  for (int p = 0; p < PP; ++p)
    v += beta[p] * (float)z[(size_t)p * NN * CC + idx];
  const float e = v > 0.f ? v : (__expf(v) - 1.f);  // ELU(alpha=1)
  red[c] = e;
  __syncthreads();
  for (int s = 128; s > 0; s >>= 1) {
    if (c < s) red[c] += red[c + s];
    __syncthreads();
  }
  if (c == 0) s_mu = red[0] * (1.f / CC);
  __syncthreads();
  const float d = e - s_mu;
  red[c] = d * d;
  __syncthreads();
  for (int s = 128; s > 0; s >>= 1) {
    if (c < s) red[c] += red[c + s];
    __syncthreads();
  }
  if (c == 0) s_var = red[0] * (1.f / CC);
  __syncthreads();
  xout[idx] = d * rsqrtf(s_var + 1e-5f) * g[c] + b[c];
}

__global__ void pool_zero_kernel(float* __restrict__ pooled,
                                 float* __restrict__ cnt) {
  const int i = blockIdx.x * 256 + threadIdx.x;
  if (i < BB * CC) pooled[i] = 0.f;
  if (i < BB) cnt[i] = 0.f;
}

__global__ __launch_bounds__(256) void pool_add_kernel(
    const float* __restrict__ x, const int* __restrict__ batch,
    float* __restrict__ pooled, float* __restrict__ cnt) {
  const int n = blockIdx.x, c = threadIdx.x;
  const int b = batch[n];
  atomicAdd(&pooled[(size_t)b * CC + c], x[(size_t)n * CC + c]);
  if (c == 0) atomicAdd(&cnt[b], 1.f);
}

__global__ __launch_bounds__(256) void pool_out_kernel(
    const float* __restrict__ pooled, const float* __restrict__ cnt,
    const float* __restrict__ linW, const float* __restrict__ linb,
    float* __restrict__ out) {
  __shared__ float red[CC];
  const int b = blockIdx.x, c = threadIdx.x;
  red[c] = pooled[(size_t)b * CC + c] * linW[c];
  __syncthreads();
  for (int s = 128; s > 0; s >>= 1) {
    if (c < s) red[c] += red[c + s];
    __syncthreads();
  }
  if (c == 0) out[b] = red[0] / fmaxf(cnt[b], 1.f) + linb[0];
}

// ============================================================================
extern "C" void kernel_launch(void* const* d_in, const int* in_sizes, int n_in,
                              void* d_out, int out_size, void* d_ws,
                              size_t ws_size, hipStream_t stream) {
  (void)in_sizes; (void)n_in; (void)out_size; (void)ws_size;
  // ---- inputs (setup_inputs order; params = jax pytree flatten:
  //      dicts by sorted key, tuples in order) ----
  const float* x_operator = (const float*)d_in[0];
  const int*   edges      = (const int*)d_in[7];   // [P,2,E]
  const int*   batch      = (const int*)d_in[8];   // [N]
  // params: conv1(W,att_dst,att_src,b,kW,kb,q) conv2(...) lin(W,b)
  //         norm1(g,b) norm2(g,b) proj(column,literal,numeral,operation,
  //         operator,predicate,table; each (W,b))
  const float* cW[2]    = {(const float*)d_in[9],  (const float*)d_in[16]};
  const float* cattd[2] = {(const float*)d_in[10], (const float*)d_in[17]};
  const float* catts[2] = {(const float*)d_in[11], (const float*)d_in[18]};
  const float* cb[2]    = {(const float*)d_in[12], (const float*)d_in[19]};
  const float* ckW[2]   = {(const float*)d_in[13], (const float*)d_in[20]};
  const float* ckb[2]   = {(const float*)d_in[14], (const float*)d_in[21]};
  const float* cq[2]    = {(const float*)d_in[15], (const float*)d_in[22]};
  const float* lin_W = (const float*)d_in[23];
  const float* lin_b = (const float*)d_in[24];
  const float* ng[2] = {(const float*)d_in[25], (const float*)d_in[27]};
  const float* nb[2] = {(const float*)d_in[26], (const float*)d_in[28]};
  const float* proj_opW = (const float*)d_in[37];  // proj['operator'] W [4,C]
  const float* proj_opb = (const float*)d_in[38];
  float* out = (float*)d_out;

  // ---- workspace carve-out (256 B aligned) ----
  char* ws = (char*)d_ws;
  size_t off = 0;
  auto carve = [&](size_t bytes) -> char* {
    char* p = ws + off;
    off = (off + bytes + 255) & ~(size_t)255;
    return p;
  };
  __bf16*  xbf    = (__bf16*)carve((size_t)NN * CC * 2);
  float*   hW     = (float*)carve((size_t)NN * CC * 4);
  float*   a_src  = (float*)carve((size_t)PP * NN * HH * 4);
  float*   a_dst  = (float*)carve((size_t)PP * NN * HH * 4);
  float*   e_st   = (float*)carve((size_t)EE * HH * 4);
  unsigned* mkey  = (unsigned*)carve((size_t)NN * HH * 4);
  float*   denom  = (float*)carve((size_t)NN * HH * 4);
  float*   outacc = (float*)carve((size_t)NN * CC * 4);
  __bf16*  zbf    = (__bf16*)carve((size_t)PP * NN * CC * 2);
  float*   xbuf   = (float*)carve((size_t)NN * CC * 4);
  __bf16*  Wbf[2]  = {(__bf16*)carve(CC * CC * 2), (__bf16*)carve(CC * CC * 2)};
  __bf16*  kWbf[2] = {(__bf16*)carve(CC * CC * 2), (__bf16*)carve(CC * CC * 2)};
  float*   wbar   = (float*)carve(PP * 4);
  float*   beta   = (float*)carve(PP * 4);
  float*   pooled = (float*)carve((size_t)BB * CC * 4);
  float*   cnt    = (float*)carve(BB * 4);

  const int NC_BLK = (NN * CC) / 256;                 // 100000 blocks
  const int W1 = (NN / 16) * (CC / 64);               // gemm1 wave tiles
  const int W2 = ((PP * NN) / 16) * (CC / 64);        // gemm2 wave tiles

  // ---- 0. projection of operator features -> xbuf ----
  proj_op_kernel<<<NC_BLK, 256, 0, stream>>>(x_operator, proj_opW, proj_opb, xbuf);
  // weights to bf16 once
  for (int i = 0; i < 2; ++i) {
    f32_to_bf16_kernel<<<(CC * CC) / 256, 256, 0, stream>>>(cW[i], Wbf[i],
                                                            (size_t)CC * CC);
    f32_to_bf16_kernel<<<(CC * CC) / 256, 256, 0, stream>>>(ckW[i], kWbf[i],
                                                            (size_t)CC * CC);
  }

  // ---- two HAN conv layers ----
  for (int L = 0; L < 2; ++L) {
    // h = x @ W + b  (bf16 WMMA, fp32 accumulate)
    f32_to_bf16_kernel<<<NC_BLK, 256, 0, stream>>>(xbuf, xbf, (size_t)NN * CC);
    wmma_gemm_bias<<<W1 / 8, 256, 0, stream>>>(xbf, Wbf[L], cb[L], hW, W1);
    // attention coefficients
    att_kernel<<<NN, 128, 0, stream>>>(hW, catts[L], cattd[L], a_src, a_dst);
    // per meta-path scatter softmax + message aggregation
    for (int p = 0; p < PP; ++p) {
      const int* src = edges + ((size_t)p * 2 + 0) * EE;
      const int* dst = edges + ((size_t)p * 2 + 1) * EE;
      const float* asp = a_src + (size_t)p * NN * HH;
      const float* adp = a_dst + (size_t)p * NN * HH;
      path_init_kernel<<<NC_BLK, 256, 0, stream>>>(outacc, mkey, denom);
      edge_pass1<<<(EE + 255) / 256, 256, 0, stream>>>(src, dst, asp, adp,
                                                       e_st, mkey);
      edge_pass2<<<(EE + 255) / 256, 256, 0, stream>>>(dst, e_st, mkey, denom);
      edge_pass3<<<EE, 256, 0, stream>>>(src, dst, e_st, denom, hW, outacc);
      relu_cvt_kernel<<<NC_BLK, 256, 0, stream>>>(outacc,
                                                  zbf + (size_t)p * NN * CC);
    }
    // semantic attention: wbar[p] = sum_n,c tanh((z@kW + kb)) * q  (fused)
    zero_f_kernel<<<1, 32, 0, stream>>>(wbar, PP);
    wmma_gemm_tanhq<<<W2 / 8, 256, 0, stream>>>(zbf, kWbf[L], ckb[L], cq[L],
                                                wbar, W2);
    beta_kernel<<<1, 32, 0, stream>>>(wbar, beta);
    // x = layer_norm(elu(sum_p beta_p z_p))
    combine_elu_norm<<<NN, 256, 0, stream>>>(zbf, beta, ng[L], nb[L], xbuf);
  }

  // ---- global mean pool + linear head ----
  pool_zero_kernel<<<(BB * CC + 255) / 256, 256, 0, stream>>>(pooled, cnt);
  pool_add_kernel<<<NN, 256, 0, stream>>>(xbuf, batch, pooled, cnt);
  pool_out_kernel<<<BB, 256, 0, stream>>>(pooled, cnt, lin_W, lin_b, out);
}
